// LayerK_34926674051408
// MI455X (gfx1250) — matrix operand order repaired and checked by reading, not compiled, not run on hardware
//
#include <hip/hip_runtime.h>
#include <hip/hip_bf16.h>
#include <math.h>

// ---- problem constants (from reference): F=128, H=8, C=16 ----
#define FDIM 128
#define HEADS 8
#define CH 16
#define NEG_SLOPE 0.2f
#define EPSV 1e-16f
#define SELU_SCALE 1.0507009873554805f
#define SELU_ALPHA 1.6732632423543772f

typedef _Float16 v16h __attribute__((ext_vector_type(16)));
typedef _Float16 v8h  __attribute__((ext_vector_type(8)));
typedef float    v8f  __attribute__((ext_vector_type(8)));

union V16 { v16h v; v8h h[2]; };

// order-preserving float -> uint map (so u32 atomicMax == float max)
__device__ __forceinline__ unsigned int enc_f32(float f) {
    unsigned int u = __float_as_uint(f);
    return (u & 0x80000000u) ? ~u : (u | 0x80000000u);
}
__device__ __forceinline__ float dec_f32(unsigned int u) {
    u = (u & 0x80000000u) ? (u & 0x7FFFFFFFu) : ~u;
    return __uint_as_float(u);
}
__device__ __forceinline__ float leaky(float v) { return v > 0.0f ? v : NEG_SLOPE * v; }

// ---------------- K0: init out / amax / asum ----------------
__global__ void gat_init_kernel(float* __restrict__ out, unsigned int* __restrict__ amax,
                                float* __restrict__ asum, int n_out, int n_nh) {
    int tid = blockIdx.x * blockDim.x + threadIdx.x;
    if (tid < n_out) out[tid] = 0.0f;
    if (tid < n_nh) { amax[tid] = enc_f32(-1e30f); asum[tid] = 0.0f; }
}

// ---------------- K1: x = A @ W via v_wmma_f32_16x16x32_f16 ----------------
// block = 256 threads (8 wave32), computes 64 rows x 128 cols.
__global__ void gat_gemm_wmma_kernel(const float* __restrict__ A,  // [N,128]
                                     const float* __restrict__ W,  // [128,128]
                                     float* __restrict__ X,        // [N,128]
                                     int N) {
    __shared__ __attribute__((aligned(32))) _Float16 Alds[64 * FDIM];   // 16 KB, row-major
    __shared__ __attribute__((aligned(32))) _Float16 Wlds[FDIM * FDIM]; // 32 KB, TRANSPOSED: Wlds[n*128+k]

    const int t = threadIdx.x;
    const int rowBase = blockIdx.x * 64;

    // stage A tile (64x128), guarded + converted to f16
    #pragma unroll
    for (int i = 0; i < 32; ++i) {
        int idx = t + 256 * i;                 // 8192 elems
        int r = idx >> 7, c = idx & 127;
        int gr = rowBase + r;
        float v = (gr < N) ? A[(size_t)gr * FDIM + c] : 0.0f;
        Alds[idx] = (_Float16)v;
    }
    // stage W transposed (128x128)
    #pragma unroll
    for (int i = 0; i < 64; ++i) {
        int idx = t + 256 * i;                 // 16384 elems
        int k = idx >> 7, n = idx & 127;
        Wlds[n * FDIM + k] = (_Float16)W[idx];
    }
    __syncthreads();

    const int wave = t >> 5;        // 0..7 (wave32)
    const int lane = t & 31;
    const int ln   = lane & 15;     // M for A-frag, N for B-frag
    const int h2   = lane >> 4;     // lane half
    const int sr   = wave & 3;      // row strip (16 rows each)
    const int chf  = wave >> 2;     // column half (64 cols each)

    v8f acc[4];
    #pragma unroll
    for (int tt = 0; tt < 4; ++tt) acc[tt] = (v8f){0.f,0.f,0.f,0.f,0.f,0.f,0.f,0.f};

    #pragma unroll
    for (int kc = 0; kc < 4; ++kc) {
        const int k0 = kc * 32;
        // A fragment (16x32 f16): lane half h2 holds K = k0 + 16*h2 .. +15 contiguous
        V16 a;
        a.h[0] = *(const v8h*)&Alds[(sr * 16 + ln) * FDIM + k0 + 16 * h2];
        a.h[1] = *(const v8h*)&Alds[(sr * 16 + ln) * FDIM + k0 + 16 * h2 + 8];
        #pragma unroll
        for (int tt = 0; tt < 4; ++tt) {
            // B fragment (32x16 f16): col = tile col + ln, K = k0 + 16*h2 .. +15 contiguous
            V16 b;
            b.v = *(const v16h*)&Wlds[(chf * 64 + tt * 16 + ln) * FDIM + k0 + 16 * h2];
            acc[tt] = __builtin_amdgcn_wmma_f32_16x16x32_f16(
                false, a.v, false, b.v, (short)0, acc[tt], false, false);
        }
    }

    // store: C/D layout -> VGPR r holds M = r + 8*h2, N = ln
    #pragma unroll
    for (int tt = 0; tt < 4; ++tt) {
        #pragma unroll
        for (int r = 0; r < 8; ++r) {
            int row = rowBase + sr * 16 + r + 8 * h2;
            if (row < N) X[(size_t)row * FDIM + chf * 64 + tt * 16 + ln] = acc[tt][r];
        }
    }
}

// ---------------- K2: per-(node,head) attention logits ----------------
__global__ void gat_logits_kernel(const float* __restrict__ X,
                                  const float* __restrict__ att_src,
                                  const float* __restrict__ att_dst,
                                  float* __restrict__ a_src, float* __restrict__ a_dst,
                                  int NH) {
    int tid = blockIdx.x * blockDim.x + threadIdx.x;
    if (tid >= NH) return;
    int n = tid >> 3, h = tid & 7;
    const float* xp = X + (size_t)n * FDIM + h * CH;
    const float* as = att_src + h * CH;
    const float* ad = att_dst + h * CH;
    float s = 0.0f, d = 0.0f;
    #pragma unroll
    for (int c = 0; c < CH; ++c) { float xv = xp[c]; s += xv * as[c]; d += xv * ad[c]; }
    a_src[tid] = s; a_dst[tid] = d;
}

// ---------------- K3: edge pass A — segment max (u32-encoded atomicMax) ----------------
__global__ void gat_edge_max_kernel(const int* __restrict__ ei, int E, int N,
                                    const float* __restrict__ a_src,
                                    const float* __restrict__ a_dst,
                                    unsigned int* __restrict__ amax) {
    int e = blockIdx.x * blockDim.x + threadIdx.x;
    int ET = E + N;                                  // self loops appended
    if (e >= ET) return;
    int s = (e < E) ? ei[e] : (e - E);
    int d = (e < E) ? ei[E + e] : (e - E);
    #pragma unroll
    for (int h = 0; h < HEADS; ++h) {
        float l = leaky(a_src[s * HEADS + h] + a_dst[d * HEADS + h]);
        atomicMax(&amax[d * HEADS + h], enc_f32(l));
    }
}

// ---------------- K4: edge pass B — exp-sum ----------------
__global__ void gat_edge_sum_kernel(const int* __restrict__ ei, int E, int N,
                                    const float* __restrict__ a_src,
                                    const float* __restrict__ a_dst,
                                    const unsigned int* __restrict__ amax,
                                    float* __restrict__ asum) {
    int e = blockIdx.x * blockDim.x + threadIdx.x;
    int ET = E + N;
    if (e >= ET) return;
    int s = (e < E) ? ei[e] : (e - E);
    int d = (e < E) ? ei[E + e] : (e - E);
    #pragma unroll
    for (int h = 0; h < HEADS; ++h) {
        float l = leaky(a_src[s * HEADS + h] + a_dst[d * HEADS + h]);
        float a = __expf(l - dec_f32(amax[d * HEADS + h]));
        atomicAdd(&asum[d * HEADS + h], a);
    }
}

// ---------------- K5: edge pass C — weighted scatter-aggregate ----------------
// one thread per (edge, head): gathers 16 floats of x[src], 16 fp32 atomics to out[dst]
__global__ void gat_edge_agg_kernel(const int* __restrict__ ei, int E, int N,
                                    const float* __restrict__ X,
                                    const float* __restrict__ a_src,
                                    const float* __restrict__ a_dst,
                                    const unsigned int* __restrict__ amax,
                                    const float* __restrict__ asum,
                                    float* __restrict__ out) {
    int tid = blockIdx.x * blockDim.x + threadIdx.x;
    int ET = E + N;
    if (tid >= ET * HEADS) return;
    int e = tid >> 3, h = tid & 7;
    int s = (e < E) ? ei[e] : (e - E);
    int d = (e < E) ? ei[E + e] : (e - E);
    float l = leaky(a_src[s * HEADS + h] + a_dst[d * HEADS + h]);
    float w = __expf(l - dec_f32(amax[d * HEADS + h])) / (asum[d * HEADS + h] + EPSV);
    const float4* xp = (const float4*)(X + (size_t)s * FDIM + h * CH);
    __builtin_prefetch(xp, 0, 0);                      // global_prefetch_b8
    float* op = out + (size_t)d * FDIM + h * CH;
    #pragma unroll
    for (int j = 0; j < 4; ++j) {
        float4 xv = xp[j];
        atomicAdd(&op[4 * j + 0], xv.x * w);
        atomicAdd(&op[4 * j + 1], xv.y * w);
        atomicAdd(&op[4 * j + 2], xv.z * w);
        atomicAdd(&op[4 * j + 3], xv.w * w);
    }
}

// ---------------- K6: bias + SELU (in place) ----------------
__global__ void gat_selu_kernel(float* __restrict__ out, const float* __restrict__ bias,
                                int n_out) {
    int tid = blockIdx.x * blockDim.x + threadIdx.x;
    if (tid >= n_out) return;
    float v = out[tid] + bias[tid & (FDIM - 1)];
    out[tid] = (v > 0.0f) ? SELU_SCALE * v : SELU_SCALE * SELU_ALPHA * (__expf(v) - 1.0f);
}

extern "C" void kernel_launch(void* const* d_in, const int* in_sizes, int n_in,
                              void* d_out, int out_size, void* d_ws, size_t ws_size,
                              hipStream_t stream) {
    const float* node_feat = (const float*)d_in[0];
    const int*   edge_idx  = (const int*)d_in[1];
    const float* W         = (const float*)d_in[2];
    const float* att_src   = (const float*)d_in[3];
    const float* att_dst   = (const float*)d_in[4];
    const float* bias      = (const float*)d_in[5];
    float* out = (float*)d_out;

    const int N = in_sizes[0] / FDIM;       // 50000
    const int E = in_sizes[1] / 2;          // 600000
    const int NH = N * HEADS;
    const int n_out = N * FDIM;
    const int ET = E + N;

    // workspace layout
    char* ws = (char*)d_ws;
    float*        x_ws    = (float*)ws;                         // N*128 f32
    size_t off = (size_t)n_out * sizeof(float);
    float*        a_src_w = (float*)(ws + off); off += (size_t)NH * sizeof(float);
    float*        a_dst_w = (float*)(ws + off); off += (size_t)NH * sizeof(float);
    unsigned int* amax_w  = (unsigned int*)(ws + off); off += (size_t)NH * sizeof(unsigned int);
    float*        asum_w  = (float*)(ws + off);
    (void)ws_size; (void)n_in;

    const int TB = 256;
    gat_init_kernel<<<(n_out + TB - 1) / TB, TB, 0, stream>>>(out, amax_w, asum_w, n_out, NH);
    gat_gemm_wmma_kernel<<<(N + 63) / 64, TB, 0, stream>>>(node_feat, W, x_ws, N);
    gat_logits_kernel<<<(NH + TB - 1) / TB, TB, 0, stream>>>(x_ws, att_src, att_dst,
                                                             a_src_w, a_dst_w, NH);
    gat_edge_max_kernel<<<(ET + TB - 1) / TB, TB, 0, stream>>>(edge_idx, E, N,
                                                               a_src_w, a_dst_w, amax_w);
    gat_edge_sum_kernel<<<(ET + TB - 1) / TB, TB, 0, stream>>>(edge_idx, E, N,
                                                               a_src_w, a_dst_w, amax_w, asum_w);
    gat_edge_agg_kernel<<<((ET * HEADS) + TB - 1) / TB, TB, 0, stream>>>(
        edge_idx, E, N, x_ws, a_src_w, a_dst_w, amax_w, asum_w, out);
    gat_selu_kernel<<<(n_out + TB - 1) / TB, TB, 0, stream>>>(out, bias, n_out);
}
